// Qwen3Attention_61460982006070
// MI455X (gfx1250) — matrix operand order repaired and checked by reading, not compiled, not run on hardware
//
#include <hip/hip_runtime.h>

// ---------------------------------------------------------------------------
// Qwen3 attention block for MI455X (gfx1250, wave32, WMMA bf16 path,
// async global->LDS staging on ASYNCcnt)
// ---------------------------------------------------------------------------

#define S_LEN 2048
#define HID   2560
#define NH    32
#define NKV   8
#define HD    128
#define QSZ   (NH * HD)        // 4096
#define KVSZ  (NKV * HD)       // 1024
#define QKV_W (QSZ + 2 * KVSZ) // 6144

typedef __attribute__((ext_vector_type(16))) __bf16 v16bf;
typedef __attribute__((ext_vector_type(8)))  float  v8f;
typedef __attribute__((ext_vector_type(4)))  unsigned int u32x4;
typedef __attribute__((ext_vector_type(4)))  float  f32x4;
typedef __attribute__((ext_vector_type(4)))  int    v4i;

union Frag {
    v16bf v;
    u32x4 q[2];
    unsigned short u[16];
};

__device__ __forceinline__ unsigned short f32_to_bf16(float f) {
    union { float f; unsigned int u; } x;
    x.f = f;
    unsigned int r = x.u + 0x7FFFu + ((x.u >> 16) & 1u);
    return (unsigned short)(r >> 16);
}

// ---------------------------------------------------------------------------
// Async global->LDS copy of 16 bytes per lane (GLOBAL_LOAD_ASYNC_TO_LDS_B128,
// tracked on ASYNCcnt). Builtin is declared on this toolchain with
// (v4i as1*, v4i as3*, imm offset, imm cpol) parameters.
// ---------------------------------------------------------------------------
typedef v4i __attribute__((address_space(1)))* as1_v4i;
typedef v4i __attribute__((address_space(3)))* as3_v4i;

__device__ __forceinline__ void async_cp16(const void* g, void* l) {
#if __has_builtin(__builtin_amdgcn_global_load_async_to_lds_b128)
    __builtin_amdgcn_global_load_async_to_lds_b128(
        (as1_v4i)(unsigned long long)g,
        (as3_v4i)(unsigned int)(unsigned long long)l, 0, 0);
#else
    unsigned int lofs = (unsigned int)(unsigned long long)l;
    asm volatile("global_load_async_to_lds_b128 %0, %1, off"
                 :: "v"(lofs), "v"(g) : "memory");
#endif
}

__device__ __forceinline__ void wait_async0() {
#if __has_builtin(__builtin_amdgcn_s_wait_asynccnt)
    __builtin_amdgcn_s_wait_asynccnt(0);
#else
    asm volatile("s_wait_asynccnt 0x0" ::: "memory");
#endif
}

// ---------------------------------------------------------------------------
// fp32 -> bf16 bulk convert (n must be a multiple of 4)
// ---------------------------------------------------------------------------
__global__ __launch_bounds__(256) void cvt_f32_bf16(const float* __restrict__ src,
                                                    unsigned short* __restrict__ dst,
                                                    long long n) {
    long long i = ((long long)blockIdx.x * blockDim.x + threadIdx.x) * 4;
    if (i + 3 >= n) return;
    f32x4 f = *(const f32x4*)(src + i);
    dst[i + 0] = f32_to_bf16(f.x);
    dst[i + 1] = f32_to_bf16(f.y);
    dst[i + 2] = f32_to_bf16(f.z);
    dst[i + 3] = f32_to_bf16(f.w);
}

// extract V slice from fp32 qkv, convert to bf16 [S, NKV*HD]
__global__ __launch_bounds__(256) void extract_v(const float* __restrict__ qkv,
                                                 unsigned short* __restrict__ dst) {
    long long i = ((long long)blockIdx.x * blockDim.x + threadIdx.x) * 4;
    long long s = i >> 10;          // / KVSZ
    long long c = i & 1023;
    if (s >= S_LEN) return;
    const float* p = qkv + s * QKV_W + QSZ + KVSZ + c;
    f32x4 f = *(const f32x4*)p;
    dst[i + 0] = f32_to_bf16(f.x);
    dst[i + 1] = f32_to_bf16(f.y);
    dst[i + 2] = f32_to_bf16(f.z);
    dst[i + 3] = f32_to_bf16(f.w);
}

// ---------------------------------------------------------------------------
// RMSNorm + RoPE for q (scaled by HD^-0.5) and k. One wave per (s, head) row.
// ---------------------------------------------------------------------------
__global__ __launch_bounds__(128) void norm_rope(const float* __restrict__ qkv,
                                                 int col0,
                                                 const int* __restrict__ pos,
                                                 const float* __restrict__ w,
                                                 unsigned short* __restrict__ dst,
                                                 int nh, float scale) {
    const float LN_THETA = 9.210340371976184f; // ln(10000)
    int row  = blockIdx.x * 4 + (threadIdx.x >> 5);
    int lane = threadIdx.x & 31;
    int s = row / nh;
    int h = row % nh;
    if (s >= S_LEN) return;

    const float* x = qkv + (long long)s * QKV_W + col0 + h * HD;
    int j0 = lane * 2;

    float x1a = x[j0], x1b = x[j0 + 1];
    float x2a = x[j0 + 64], x2b = x[j0 + 65];

    float ss = x1a * x1a + x1b * x1b + x2a * x2a + x2b * x2b;
    for (int m = 16; m > 0; m >>= 1) ss += __shfl_xor(ss, m, 32);
    float r = __frsqrt_rn(ss * (1.0f / (float)HD) + 1e-6f);

    x1a *= r * w[j0];      x1b *= r * w[j0 + 1];
    x2a *= r * w[j0 + 64]; x2b *= r * w[j0 + 65];

    float p  = (float)pos[s];
    float fa = p * __expf(-((float)(2 * j0)) / (float)HD * LN_THETA);
    float fb = p * __expf(-((float)(2 * (j0 + 1))) / (float)HD * LN_THETA);
    float ca = __cosf(fa), sa = __sinf(fa);
    float cb = __cosf(fb), sb = __sinf(fb);

    unsigned short* d = dst + ((long long)s * nh + h) * HD;
    d[j0]      = f32_to_bf16((x1a * ca - x2a * sa) * scale);
    d[j0 + 1]  = f32_to_bf16((x1b * cb - x2b * sb) * scale);
    d[j0 + 64] = f32_to_bf16((x2a * ca + x1a * sa) * scale);
    d[j0 + 65] = f32_to_bf16((x2b * cb + x1b * sb) * scale);
}

// ---------------------------------------------------------------------------
// bf16 GEMM, C[M,N] = A[M,K] * B[N,K]^T (both operands K-major), fp32 accum.
// 128x128 block tile, BK=32, 8 waves of 32x64, v_wmma_f32_16x16x32_bf16.
// Double-buffered LDS staging on the async global->LDS pipe (ASYNCcnt).
// ---------------------------------------------------------------------------
#define LK 48  // padded LDS row stride (bf16 elements)

__global__ __launch_bounds__(256) void gemm_bf16_nt(const unsigned short* __restrict__ A,
                                                    const unsigned short* __restrict__ B,
                                                    float* __restrict__ C,
                                                    int M, int N, int K) {
    __shared__ unsigned short As[2][128 * LK];
    __shared__ unsigned short Bs[2][128 * LK];

    int m0 = blockIdx.y * 128;
    int n0 = blockIdx.x * 128;
    int t = threadIdx.x;
    int lane = t & 31, wave = t >> 5;
    int lmod = lane & 15, lhalf = lane >> 4;
    int wm = (wave >> 1) * 32;   // 4 wave-rows of 32
    int wn = (wave & 1) * 64;    // 2 wave-cols of 64
    int lrow = t >> 1, lseg = (t & 1) * 16;

    // stage one 128x32 tile of A and of B into LDS buffer `buf` (async)
    auto stage = [&](int buf, int k0) {
        const unsigned short* ag = A + (long long)(m0 + lrow) * K + k0 + lseg;
        const unsigned short* bg = B + (long long)(n0 + lrow) * K + k0 + lseg;
        unsigned short* as = &As[buf][lrow * LK + lseg];
        unsigned short* bs = &Bs[buf][lrow * LK + lseg];
        async_cp16(ag,     as);
        async_cp16(ag + 8, as + 8);
        async_cp16(bg,     bs);
        async_cp16(bg + 8, bs + 8);
    };

    v8f acc[2][4];
    v8f zero = {};
    for (int mi = 0; mi < 2; mi++)
        for (int ni = 0; ni < 4; ni++) acc[mi][ni] = zero;

    stage(0, 0);
    wait_async0();
    __syncthreads();

    int cur = 0;
    for (int k0 = 0; k0 < K; k0 += 32) {
        // issue async copies for the next tile while computing this one
        if (k0 + 32 < K) {
            stage(cur ^ 1, k0 + 32);
            __builtin_prefetch(A + (long long)(m0 + lrow) * K + k0 + 64 + lseg, 0, 1);
            __builtin_prefetch(B + (long long)(n0 + lrow) * K + k0 + 64 + lseg, 0, 1);
        }

        // A fragments (16x32 bf16, ISA A-layout: lane m = lane%16, koff = (lane/16)*8)
        Frag af[2];
        int koff = lhalf * 8;
        for (int mi = 0; mi < 2; mi++) {
            int mr = wm + mi * 16 + lmod;
            af[mi].q[0] = *(const u32x4*)&As[cur][mr * LK + koff];
            af[mi].q[1] = *(const u32x4*)&As[cur][mr * LK + 16 + koff];
        }
        // B fragments (32x16 bf16, lane n = lane%16, koff = (lane/16)*8, same k order as A)
        int kb16 = lhalf * 8;
        for (int ni = 0; ni < 4; ni++) {
            Frag bf;
            int nr = wn + ni * 16 + lmod;
            bf.q[0] = *(const u32x4*)&Bs[cur][nr * LK + kb16];
            bf.q[1] = *(const u32x4*)&Bs[cur][nr * LK + kb16 + 16];
            for (int mi = 0; mi < 2; mi++)
                acc[mi][ni] = __builtin_amdgcn_wmma_f32_16x16x32_bf16(
                    false, af[mi].v, false, bf.v, (short)0, acc[mi][ni], false, false);
        }

        wait_async0();      // next tile landed in LDS (this wave's copies)
        __syncthreads();    // ... and everyone else's
        cur ^= 1;
    }

    // C layout: lane holds col n = lane%16, rows m = vgpr + 8*(lane/16)
    for (int mi = 0; mi < 2; mi++)
        for (int ni = 0; ni < 4; ni++)
            for (int r = 0; r < 8; r++) {
                int m = m0 + wm + mi * 16 + lhalf * 8 + r;
                int n = n0 + wn + ni * 16 + lmod;
                C[(long long)m * N + n] = acc[mi][ni][r];
            }
}

// ---------------------------------------------------------------------------
// Flash attention (causal, GQA 4:1). Block = 64 queries x 1 head, 4 waves.
// Each wave: 16 query rows; KV processed in blocks of 32 keys.
// QK^T and PV both on v_wmma_f32_16x16x32_bf16. Online softmax in registers.
// V tile staged through the async global->LDS pipe.
// ---------------------------------------------------------------------------
__global__ __launch_bounds__(128) void flash_attn(const unsigned short* __restrict__ Q,
                                                  const unsigned short* __restrict__ Kt,
                                                  const unsigned short* __restrict__ V,
                                                  float* __restrict__ O) {
    __shared__ unsigned short vs[32][136];   // V tile: 32 keys x 128 dims (padded)
    __shared__ unsigned short ps[4][16][40]; // per-wave P staging (16x32, padded)

    int h  = blockIdx.y;
    int hk = h >> 2;
    int qb = blockIdx.x * 64;
    int wave = threadIdx.x >> 5, lane = threadIdx.x & 31;
    int lmod = lane & 15, lhalf = lane >> 4;
    int qm0 = qb + wave * 16;
    int qmax_w = qm0 + 15;

    // Preload Q fragments: 16 rows x 128 dims = 4 A-frags (scale pre-folded)
    Frag qf[4];
    {
        const unsigned short* qrow = Q + ((long long)(qm0 + lmod) * NH + h) * HD;
        int koff = lhalf * 8;
        for (int kc = 0; kc < 4; kc++) {
            qf[kc].q[0] = *(const u32x4*)(qrow + kc * 32 + koff);
            qf[kc].q[1] = *(const u32x4*)(qrow + kc * 32 + 16 + koff);
        }
    }

    float mrow[8], lrow[8];
    v8f o[8];
    v8f zero = {};
    for (int r = 0; r < 8; r++) { mrow[r] = -1e30f; lrow[r] = 0.0f; }
    for (int d = 0; d < 8; d++) o[d] = zero;

    int nkb = (qb + 63) / 32 + 1;   // all waves run same trip count (barriers!)
    for (int kb = 0; kb < nkb; kb++) {
        int kn0 = kb * 32;

        // async V tile stage (shared by all 4 waves of this head)
        {
            int r = threadIdx.x >> 2;
            int cs = (threadIdx.x & 3) * 32;
            const unsigned short* vrow = V + ((long long)(kn0 + r) * NKV + hk) * HD + cs;
            unsigned short* d = &vs[r][cs];
            async_cp16(vrow,      d);
            async_cp16(vrow + 8,  d + 8);
            async_cp16(vrow + 16, d + 16);
            async_cp16(vrow + 24, d + 24);
        }
        wait_async0();
        __syncthreads();

        if (kn0 <= qmax_w) {   // causal cutoff for this wave
            // ----- S = Q K^T (two 16x16 score tiles, K-dim = 128 = 4 chunks) -----
            v8f s0 = zero, s1 = zero;
            int kb16 = lhalf * 8;
            for (int kc = 0; kc < 4; kc++) {
                Frag b0, b1;
                const unsigned short* k0p =
                    Kt + ((long long)(kn0 + lmod) * NKV + hk) * HD + kc * 32 + kb16;
                const unsigned short* k1p =
                    Kt + ((long long)(kn0 + 16 + lmod) * NKV + hk) * HD + kc * 32 + kb16;
                b0.q[0] = *(const u32x4*)k0p; b0.q[1] = *(const u32x4*)(k0p + 16);
                b1.q[0] = *(const u32x4*)k1p; b1.q[1] = *(const u32x4*)(k1p + 16);
                s0 = __builtin_amdgcn_wmma_f32_16x16x32_bf16(false, qf[kc].v, false, b0.v,
                                                             (short)0, s0, false, false);
                s1 = __builtin_amdgcn_wmma_f32_16x16x32_bf16(false, qf[kc].v, false, b1.v,
                                                             (short)0, s1, false, false);
            }

            // ----- causal mask + online softmax (row stats via 16-lane shuffles) -----
            float p0[8], p1[8], fac[8];
            for (int r = 0; r < 8; r++) {
                int qi = qm0 + lhalf * 8 + r;
                float v0 = s0[r]; if (kn0 + lmod > qi)      v0 = -1e30f;
                float v1 = s1[r]; if (kn0 + 16 + lmod > qi) v1 = -1e30f;
                float cm = fmaxf(v0, v1);
                for (int d = 1; d < 16; d <<= 1) cm = fmaxf(cm, __shfl_xor(cm, d, 32));
                float mn = fmaxf(mrow[r], cm);
                float f  = __expf(mrow[r] - mn);
                float e0 = __expf(v0 - mn);
                float e1 = __expf(v1 - mn);
                float cs = e0 + e1;
                for (int d = 1; d < 16; d <<= 1) cs += __shfl_xor(cs, d, 32);
                lrow[r] = lrow[r] * f + cs;
                mrow[r] = mn;
                fac[r] = f; p0[r] = e0; p1[r] = e1;
            }
            for (int d = 0; d < 8; d++)
                for (int r = 0; r < 8; r++) o[d][r] *= fac[r];

            // ----- restripe P (C-layout -> A-layout) through per-wave LDS tile -----
            for (int r = 0; r < 8; r++) {
                int m = lhalf * 8 + r;
                ps[wave][m][lmod]      = f32_to_bf16(p0[r]);
                ps[wave][m][16 + lmod] = f32_to_bf16(p1[r]);
            }
            Frag pf;
            {
                int koff = lhalf * 8;
                pf.q[0] = *(const u32x4*)&ps[wave][lmod][koff];
                pf.q[1] = *(const u32x4*)&ps[wave][lmod][16 + koff];
            }

            // ----- O += P V : 8 dim-chunks of 16 -----
            for (int d8 = 0; d8 < 8; d8++) {
                Frag vf;
                for (int e = 0; e < 16; e++)
                    vf.u[e] = vs[kb16 + (e & 7) + ((e >> 3) << 4)][d8 * 16 + lmod];
                o[d8] = __builtin_amdgcn_wmma_f32_16x16x32_bf16(false, pf.v, false, vf.v,
                                                                (short)0, o[d8], false, false);
            }
        }
        __syncthreads();
    }

    // epilogue: normalize and store fp32 [S, NH*HD]
    for (int d8 = 0; d8 < 8; d8++)
        for (int r = 0; r < 8; r++) {
            int qi = qm0 + lhalf * 8 + r;
            O[(long long)qi * QSZ + h * HD + d8 * 16 + lmod] = o[d8][r] / lrow[r];
        }
}

// ---------------------------------------------------------------------------
// Host-side orchestration
// ---------------------------------------------------------------------------
extern "C" void kernel_launch(void* const* d_in, const int* in_sizes, int n_in,
                              void* d_out, int out_size, void* d_ws, size_t ws_size,
                              hipStream_t stream) {
    const int* positions   = (const int*)d_in[0];
    const float* hidden    = (const float*)d_in[1];
    const float* w_qkv     = (const float*)d_in[2];
    const float* w_o       = (const float*)d_in[3];
    const float* q_norm_w  = (const float*)d_in[4];
    const float* k_norm_w  = (const float*)d_in[5];
    float* out             = (float*)d_out;

    // workspace carving
    size_t off = 0;
    char* base = (char*)d_ws;
    auto alloc = [&](size_t bytes) -> void* {
        void* p = base + off;
        off += (bytes + 255) & ~(size_t)255;
        return p;
    };
    unsigned short* hid_bf  = (unsigned short*)alloc((size_t)S_LEN * HID * 2);
    unsigned short* wqkv_bf = (unsigned short*)alloc((size_t)QKV_W * HID * 2);
    float*          qkv_f   = (float*)alloc((size_t)S_LEN * QKV_W * 4);
    unsigned short* q_bf    = (unsigned short*)alloc((size_t)S_LEN * QSZ * 2);
    unsigned short* k_bf    = (unsigned short*)alloc((size_t)S_LEN * KVSZ * 2);
    unsigned short* v_bf    = (unsigned short*)alloc((size_t)S_LEN * KVSZ * 2);
    float*          attn_f  = (float*)alloc((size_t)S_LEN * QSZ * 4);
    unsigned short* attn_bf = (unsigned short*)alloc((size_t)S_LEN * QSZ * 2);
    unsigned short* wo_bf   = (unsigned short*)alloc((size_t)HID * QSZ * 2);

    const float SCALE = 0.08838834764831845f; // 128^-0.5

    // 1) bf16 conversions of GEMM operands
    {
        long long n1 = (long long)S_LEN * HID;
        cvt_f32_bf16<<<dim3((unsigned)(n1 / 1024)), 256, 0, stream>>>(hidden, hid_bf, n1);
        long long n2 = (long long)QKV_W * HID;
        cvt_f32_bf16<<<dim3((unsigned)(n2 / 1024)), 256, 0, stream>>>(w_qkv, wqkv_bf, n2);
        long long n3 = (long long)HID * QSZ;
        cvt_f32_bf16<<<dim3((unsigned)(n3 / 1024)), 256, 0, stream>>>(w_o, wo_bf, n3);
    }

    // 2) QKV projection: [S,HID] x [QKV_W,HID]^T -> fp32 [S,QKV_W]
    gemm_bf16_nt<<<dim3(QKV_W / 128, S_LEN / 128), 256, 0, stream>>>(
        hid_bf, wqkv_bf, qkv_f, S_LEN, QKV_W, HID);

    // 3) RMSNorm + RoPE (q scaled by HD^-0.5), extract V
    norm_rope<<<dim3(S_LEN * NH / 4), 128, 0, stream>>>(
        qkv_f, 0, positions, q_norm_w, q_bf, NH, SCALE);
    norm_rope<<<dim3(S_LEN * NKV / 4), 128, 0, stream>>>(
        qkv_f, QSZ, positions, k_norm_w, k_bf, NKV, 1.0f);
    extract_v<<<dim3((S_LEN * KVSZ) / 1024), 256, 0, stream>>>(qkv_f, v_bf);

    // 4) causal GQA flash attention -> fp32 [S, QSZ]
    flash_attn<<<dim3(S_LEN / 64, NH), 128, 0, stream>>>(q_bf, k_bf, v_bf, attn_f);

    // 5) output projection: [S,QSZ] x [HID,QSZ]^T -> fp32 [S,HID]
    {
        long long n4 = (long long)S_LEN * QSZ;
        cvt_f32_bf16<<<dim3((unsigned)(n4 / 1024)), 256, 0, stream>>>(attn_f, attn_bf, n4);
    }
    gemm_bf16_nt<<<dim3(HID / 128, S_LEN / 128), 256, 0, stream>>>(
        attn_bf, wo_bf, out, S_LEN, HID, QSZ);
}
